// KalmanVAE_29411936043283
// MI455X (gfx1250) — matrix operand order repaired and checked by reading, not compiled, not run on hardware
//
#include <hip/hip_runtime.h>
#include <hip/hip_bf16.h>
#include <math.h>

typedef __attribute__((ext_vector_type(16))) _Float16 v16h;
typedef __attribute__((ext_vector_type(8)))  float    v8f;

#define B_   1024
#define T_   256
#define HID_ 50
#define G4_  200   // 4*HID
#define NP_  208   // G4 padded to 13*16
#define KP_  64    // HID padded to 2*32
#define NT_  13    // N tiles of 16

__device__ __forceinline__ float sigmf(float x) { return 1.f / (1.f + __expf(-x)); }

// Inverse of this file's contiguous A-operand k order (K = s + 16g):
// for k-within-chunk kk: lane-half g and slot s
__device__ __forceinline__ int haG(int k) { return (((k & 31) >> 4) & 1); }
__device__ __forceinline__ int haS(int k) { int kk = k & 31; return kk & 15; }

// ---------------------------------------------------------------------------
// Kernel 1: layer-0 input projection (K=2, VALU), written in C-tile layout:
// xprojT[((btile*T + t)*13 + tn)*32 + lane][8]  with lane=16g+m, r: M=r+8g,N=m
// ---------------------------------------------------------------------------
__global__ void xproj0_kernel(const float* __restrict__ obs,
                              const float* __restrict__ Wih0,
                              const float* __restrict__ bih0,
                              const float* __restrict__ bhh0,
                              const float* __restrict__ a1,
                              float* __restrict__ xprojT) {
  size_t idx = (size_t)blockIdx.x * blockDim.x + threadIdx.x;
  size_t total = (size_t)B_ * T_ * NP_;
  if (idx >= total) return;
  int n = (int)(idx % NP_);
  size_t bt = idx / NP_;
  int t = (int)(bt % T_);
  int b = (int)(bt / T_);
  float v = 0.f;
  if (n < G4_) {
    float x0 = (t == 0) ? a1[0] : obs[((size_t)b * T_ + (t - 1)) * 2 + 0];
    float x1 = (t == 0) ? a1[1] : obs[((size_t)b * T_ + (t - 1)) * 2 + 1];
    v = bih0[n] + bhh0[n] + x0 * Wih0[n * 2 + 0] + x1 * Wih0[n * 2 + 1];
  }
  int btile = b >> 4, M = b & 15;
  int r = M & 7, g = M >> 3;
  int tn = n >> 4, m = n & 15;
  size_t o = ((((size_t)btile * T_ + t) * NT_ + tn) * 32 + (g * 16 + m)) * 8 + r;
  xprojT[o] = v;
}

// ---------------------------------------------------------------------------
// LSTM recurrence: one workgroup (4 waves) owns 16 batch rows for ALL steps.
// W_hh^T lives in registers as WMMA B operands; h/c live in LDS.
// xproj tiles are register double-buffered (load t+1 during step t) and
// prefetched (t+2) via global_prefetch to hide global latency in the
// sequential loop.
// ---------------------------------------------------------------------------
template <int LAYER>
__global__ void lstm_rec_kernel(const float* __restrict__ Whh,
                                const float* __restrict__ xprojT,
                                _Float16* __restrict__ h0out,   // layer0: A-layout tiles
                                const float* __restrict__ Walpha,
                                const float* __restrict__ balpha,
                                float* __restrict__ wsm) {
  __shared__ alignas(32) _Float16 sWB[NT_][2][32][16];  // B operands
  __shared__ alignas(32) float    sGateT[NT_][32][8];   // C/D tiles
  __shared__ alignas(32) _Float16 sHA[2][32][16];       // A operand (h state)
  __shared__ float sC[16][HID_];
  __shared__ float sWa[3][HID_];
  __shared__ float sBa[3];

  const int tid = threadIdx.x;
  const int btile = blockIdx.x;

  // Stage W_hh^T directly in WMMA B-operand layout
  for (int i = tid; i < NT_ * 2 * 32 * 16; i += 128) {
    int s = i & 15;
    int ln = (i >> 4) & 31;
    int kc = (i >> 9) & 1;
    int tn = i >> 10;
    int k = kc * 32 + ((ln >> 4) << 4) + s;
    int n = tn * 16 + (ln & 15);
    sWB[tn][kc][ln][s] =
        (k < HID_ && n < G4_) ? (_Float16)Whh[n * HID_ + k] : (_Float16)0.f;
  }
  for (int i = tid; i < 2 * 32 * 16; i += 128)
    ((_Float16*)sHA)[i] = (_Float16)0.f;
  for (int i = tid; i < 16 * HID_; i += 128) sC[i / HID_][i % HID_] = 0.f;
  if (LAYER == 1) {
    for (int i = tid; i < 3 * HID_; i += 128) sWa[i / HID_][i % HID_] = Walpha[i];
    if (tid < 3) sBa[tid] = balpha[tid];
  }
  __syncthreads();

  const int lane = tid & 31;
  const int wv = tid >> 5;

  // Preload time-invariant B operands into registers (vector LDS loads)
  v16h Breg[4][2];
#pragma unroll
  for (int i = 0; i < 4; ++i) {
    int tn = wv + 4 * i;
    if (tn < NT_) {
      Breg[i][0] = *(const v16h*)&sWB[tn][0][lane][0];
      Breg[i][1] = *(const v16h*)&sWB[tn][1][lane][0];
    }
  }

  // xproj tile base address helper (C-tile layout)
  auto xbase = [&](int t, int tn) -> size_t {
    return ((((size_t)btile * T_ + t) * NT_ + tn) * 32 + lane) * 8;
  };

  // Register double-buffer: preload step 0's accumulator tiles
  v8f accBuf[4];
#pragma unroll
  for (int i = 0; i < 4; ++i) {
    int tn = wv + 4 * i;
    if (tn < NT_) accBuf[i] = *(const v8f*)(xprojT + xbase(0, tn));
  }

  for (int t = 0; t < T_; ++t) {
    // A operand: two contiguous 32B LDS reads
    v16h A0 = *(const v16h*)&sHA[0][lane][0];
    v16h A1 = *(const v16h*)&sHA[1][lane][0];

    // Issue next step's accumulator loads early (latency hidden behind
    // WMMAs + elementwise phase + barriers); prefetch t+2 into WGP cache.
    v8f accNext[4];
    if (t + 1 < T_) {
#pragma unroll
      for (int i = 0; i < 4; ++i) {
        int tn = wv + 4 * i;
        if (tn < NT_) {
          accNext[i] = *(const v8f*)(xprojT + xbase(t + 1, tn));
          if (t + 2 < T_) __builtin_prefetch(xprojT + xbase(t + 2, tn), 0, 3);
        }
      }
    }

#pragma unroll
    for (int i = 0; i < 4; ++i) {
      int tn = wv + 4 * i;
      if (tn < NT_) {
        v8f acc = accBuf[i];  // accumulator = xproj + biases
        acc = __builtin_amdgcn_wmma_f32_16x16x32_f16(false, A0, false,
                                                     Breg[i][0], (short)0, acc,
                                                     false, false);
        acc = __builtin_amdgcn_wmma_f32_16x16x32_f16(false, A1, false,
                                                     Breg[i][1], (short)0, acc,
                                                     false, false);
        *(v8f*)&sGateT[tn][lane][0] = acc;  // vector LDS store
      }
    }
    __syncthreads();

    // elementwise LSTM cell (gate order i, f, g, o); gates read from C-tiles
    for (int idx = tid; idx < 16 * HID_; idx += 128) {
      int row = idx / HID_;
      int j = idx - row * HID_;
#define GATE(col) sGateT[(col) >> 4][((row >> 3) << 4) | ((col) & 15)][row & 7]
      float gi = GATE(j);
      float gf = GATE(HID_ + j);
      float gg = GATE(2 * HID_ + j);
      float go = GATE(3 * HID_ + j);
#undef GATE
      float c = sC[row][j];
      c = sigmf(gf) * c + sigmf(gi) * tanhf(gg);
      sC[row][j] = c;
      float h = sigmf(go) * tanhf(c);
      // write h into A-operand layout (pad slots stay zero forever)
      sHA[j >> 5][(haG(j) << 4) + row][haS(j)] = (_Float16)h;
    }
    __syncthreads();

    if (LAYER == 0) {
      // copy the A-layout h tile to global as 16B chunks (b128 stores)
      const uint4* src = (const uint4*)&sHA[0][0][0];
      uint4* dst = (uint4*)(h0out + ((size_t)btile * T_ + t) * 1024);
      dst[tid] = src[tid];
    }
    if (LAYER == 1) {
      if (tid < 16) {
        int row = tid;
        float lg[3];
#pragma unroll
        for (int k = 0; k < 3; ++k) {
          float a = sBa[k];
          for (int j = 0; j < HID_; ++j)
            a += (float)sHA[j >> 5][(haG(j) << 4) + row][haS(j)] * sWa[k][j];
          lg[k] = a;
        }
        float mx = fmaxf(lg[0], fmaxf(lg[1], lg[2]));
        float e0 = __expf(lg[0] - mx), e1 = __expf(lg[1] - mx), e2 = __expf(lg[2] - mx);
        float inv = 1.f / (e0 + e1 + e2);
        size_t base = ((size_t)(btile * 16 + row) * T_ + t) * 3;
        wsm[base + 0] = e0 * inv;
        wsm[base + 1] = e1 * inv;
        wsm[base + 2] = e2 * inv;
      }
    }
    __syncthreads();

    if (t + 1 < T_) {
#pragma unroll
      for (int i = 0; i < 4; ++i) accBuf[i] = accNext[i];
    }
  }
}

// ---------------------------------------------------------------------------
// Kernel 3: dense WMMA GEMM  xprojT(tiles) = h0A(tiles) @ W_ih1^T + biases
// A: contiguous global b128 loads; D: contiguous global b128 stores.
// ---------------------------------------------------------------------------
__global__ void xproj1_gemm_kernel(const _Float16* __restrict__ h0A,
                                   const float* __restrict__ Wih1,
                                   const float* __restrict__ bih1,
                                   const float* __restrict__ bhh1,
                                   float* __restrict__ xprojT) {
  __shared__ alignas(32) _Float16 sWB[NT_][2][32][16];
  __shared__ float sBias[NP_];
  const int tid = threadIdx.x;
  for (int i = tid; i < NT_ * 2 * 32 * 16; i += 128) {
    int s = i & 15;
    int ln = (i >> 4) & 31;
    int kc = (i >> 9) & 1;
    int tn = i >> 10;
    int k = kc * 32 + ((ln >> 4) << 4) + s;
    int n = tn * 16 + (ln & 15);
    sWB[tn][kc][ln][s] =
        (k < HID_ && n < G4_) ? (_Float16)Wih1[n * HID_ + k] : (_Float16)0.f;
  }
  for (int i = tid; i < NP_; i += 128)
    sBias[i] = (i < G4_) ? (bih1[i] + bhh1[i]) : 0.f;
  __syncthreads();

  const int lane = tid & 31, wv = tid >> 5;
  const int m = lane & 15;
  const size_t mt = (size_t)blockIdx.x * 4 + wv;

  const v16h A0 = *(const v16h*)(h0A + (mt * 2 + 0) * 512 + lane * 16);
  const v16h A1 = *(const v16h*)(h0A + (mt * 2 + 1) * 512 + lane * 16);

  for (int tn = 0; tn < NT_; ++tn) {
    v16h B0 = *(const v16h*)&sWB[tn][0][lane][0];
    v16h B1 = *(const v16h*)&sWB[tn][1][lane][0];
    float bias = sBias[tn * 16 + m];
    v8f acc;
#pragma unroll
    for (int r = 0; r < 8; ++r) acc[r] = bias;
    acc = __builtin_amdgcn_wmma_f32_16x16x32_f16(false, A0, false, B0, (short)0,
                                                 acc, false, false);
    acc = __builtin_amdgcn_wmma_f32_16x16x32_f16(false, A1, false, B1, (short)0,
                                                 acc, false, false);
    *(v8f*)(xprojT + ((mt * NT_ + tn) * 32 + lane) * 8) = acc;
  }
}

// ---------------------------------------------------------------------------
// Kernel 4: Kalman forward filter (f64, one thread per batch element)
// ---------------------------------------------------------------------------
__global__ void kalman_filter_kernel(const float* __restrict__ obs,
                                     const float* __restrict__ wsm,
                                     const float* __restrict__ Ak,
                                     const float* __restrict__ Ck,
                                     double* __restrict__ muf, double* __restrict__ sigf,
                                     double* __restrict__ mup, double* __restrict__ sigp) {
  __shared__ float sA[3][16];
  __shared__ float sCk[3][8];
  int tid = threadIdx.x;
  if (tid < 48) sA[tid >> 4][tid & 15] = Ak[tid];
  if (tid < 24) sCk[tid >> 3][tid & 7] = Ck[tid];
  __syncthreads();
  int b = blockIdx.x * blockDim.x + tid;
  if (b >= B_) return;

  double mu_pred[4] = {0, 0, 0, 0};
  double sig_pred[16];
#pragma unroll
  for (int i = 0; i < 16; ++i) sig_pred[i] = 0.0;
  sig_pred[0] = sig_pred[5] = sig_pred[10] = sig_pred[15] = 20.0;
  double mu_prev[4] = {0, 0, 0, 0};

  double At[16];
  {
    const float* w = &wsm[((size_t)b * T_) * 3];
#pragma unroll
    for (int i = 0; i < 16; ++i)
      At[i] = (double)(w[0] * sA[0][i] + w[1] * sA[1][i] + w[2] * sA[2][i]);
  }

  for (int t = 0; t < T_; ++t) {
    const float* w = &wsm[((size_t)b * T_ + t) * 3];
    double Ct[8];
#pragma unroll
    for (int i = 0; i < 8; ++i)
      Ct[i] = (double)(w[0] * sCk[0][i] + w[1] * sCk[1][i] + w[2] * sCk[2][i]);

    double M1[8];
#pragma unroll
    for (int i = 0; i < 2; ++i)
#pragma unroll
      for (int j = 0; j < 4; ++j) {
        double s = 0;
#pragma unroll
        for (int l = 0; l < 4; ++l) s += Ct[i * 4 + l] * sig_pred[l * 4 + j];
        M1[i * 4 + j] = s;
      }
    double S00 = 0.03, S01 = 0, S10 = 0, S11 = 0.03;
#pragma unroll
    for (int l = 0; l < 4; ++l) {
      S00 += M1[l] * Ct[l];
      S01 += M1[l] * Ct[4 + l];
      S10 += M1[4 + l] * Ct[l];
      S11 += M1[4 + l] * Ct[4 + l];
    }
    double idet = 1.0 / (S00 * S11 - S01 * S10);
    double i00 = S11 * idet, i01 = -S01 * idet, i10 = -S10 * idet, i11 = S00 * idet;

    double T1[8];
#pragma unroll
    for (int i = 0; i < 4; ++i) {
      double a = 0, bb = 0;
#pragma unroll
      for (int l = 0; l < 4; ++l) {
        a += sig_pred[i * 4 + l] * Ct[l];
        bb += sig_pred[i * 4 + l] * Ct[4 + l];
      }
      T1[i * 2 + 0] = a;
      T1[i * 2 + 1] = bb;
    }
    double Kg[8];
#pragma unroll
    for (int i = 0; i < 4; ++i) {
      Kg[i * 2 + 0] = T1[i * 2] * i00 + T1[i * 2 + 1] * i10;
      Kg[i * 2 + 1] = T1[i * 2] * i01 + T1[i * 2 + 1] * i11;
    }
    double Amu[4];
#pragma unroll
    for (int i = 0; i < 4; ++i) {
      double s = 0;
#pragma unroll
      for (int l = 0; l < 4; ++l) s += At[i * 4 + l] * mu_prev[l];
      Amu[i] = s;
    }
    double y0 = (double)obs[((size_t)b * T_ + t) * 2 + 0];
    double y1 = (double)obs[((size_t)b * T_ + t) * 2 + 1];
    double r0 = y0 - (Ct[0] * Amu[0] + Ct[1] * Amu[1] + Ct[2] * Amu[2] + Ct[3] * Amu[3]);
    double r1 = y1 - (Ct[4] * Amu[0] + Ct[5] * Amu[1] + Ct[6] * Amu[2] + Ct[7] * Amu[3]);
    double mu_f[4];
#pragma unroll
    for (int i = 0; i < 4; ++i)
      mu_f[i] = Amu[i] + Kg[i * 2] * r0 + Kg[i * 2 + 1] * r1;

    double IM[16];
#pragma unroll
    for (int i = 0; i < 4; ++i)
#pragma unroll
      for (int j = 0; j < 4; ++j)
        IM[i * 4 + j] = ((i == j) ? 1.0 : 0.0) - (Kg[i * 2] * Ct[j] + Kg[i * 2 + 1] * Ct[4 + j]);
    double sig_f[16];
#pragma unroll
    for (int i = 0; i < 4; ++i)
#pragma unroll
      for (int j = 0; j < 4; ++j) {
        double s = 0;
#pragma unroll
        for (int l = 0; l < 4; ++l) s += IM[i * 4 + l] * sig_pred[l * 4 + j];
        sig_f[i * 4 + j] = s;
      }

    size_t o4 = ((size_t)t * B_ + b) * 4;
    size_t o16 = ((size_t)t * B_ + b) * 16;
#pragma unroll
    for (int i = 0; i < 4; ++i) { muf[o4 + i] = mu_f[i]; mup[o4 + i] = mu_pred[i]; }
#pragma unroll
    for (int i = 0; i < 16; ++i) { sigf[o16 + i] = sig_f[i]; sigp[o16 + i] = sig_pred[i]; }

    double At1[16];
    if (t == T_ - 1) {
#pragma unroll
      for (int i = 0; i < 16; ++i) At1[i] = ((i % 5) == 0) ? 1.0 : 0.0;
    } else {
      const float* w1 = &wsm[((size_t)b * T_ + t + 1) * 3];
#pragma unroll
      for (int i = 0; i < 16; ++i)
        At1[i] = (double)(w1[0] * sA[0][i] + w1[1] * sA[1][i] + w1[2] * sA[2][i]);
    }
#pragma unroll
    for (int i = 0; i < 4; ++i) {
      double s = 0;
#pragma unroll
      for (int l = 0; l < 4; ++l) s += At1[i * 4 + l] * mu_f[l];
      mu_pred[i] = s;
    }
    double AS[16];
#pragma unroll
    for (int i = 0; i < 4; ++i)
#pragma unroll
      for (int j = 0; j < 4; ++j) {
        double s = 0;
#pragma unroll
        for (int l = 0; l < 4; ++l) s += At1[i * 4 + l] * sig_f[l * 4 + j];
        AS[i * 4 + j] = s;
      }
#pragma unroll
    for (int i = 0; i < 4; ++i)
#pragma unroll
      for (int j = 0; j < 4; ++j) {
        double s = (i == j) ? 0.08 : 0.0;
#pragma unroll
        for (int l = 0; l < 4; ++l) s += AS[i * 4 + l] * At1[j * 4 + l];
        sig_pred[i * 4 + j] = s;
      }
#pragma unroll
    for (int i = 0; i < 4; ++i) mu_prev[i] = mu_f[i];
#pragma unroll
    for (int i = 0; i < 16; ++i) At[i] = At1[i];
  }
}

// ---------------------------------------------------------------------------
// Kernel 5: RTS smoother (reverse, f64). 4x4 inverse via Cholesky (SPD).
// ---------------------------------------------------------------------------
__global__ void kalman_smooth_kernel(const float* __restrict__ wsm,
                                     const float* __restrict__ Ak,
                                     const double* __restrict__ muf, const double* __restrict__ sigf,
                                     const double* __restrict__ mup, const double* __restrict__ sigp,
                                     float* __restrict__ out) {
  __shared__ float sA[3][16];
  int tid = threadIdx.x;
  if (tid < 48) sA[tid >> 4][tid & 15] = Ak[tid];
  __syncthreads();
  int b = blockIdx.x * blockDim.x + tid;
  if (b >= B_) return;

  double mu_s[4], sig_s[16];
  {
    size_t o4 = ((size_t)(T_ - 1) * B_ + b) * 4;
    size_t o16 = ((size_t)(T_ - 1) * B_ + b) * 16;
#pragma unroll
    for (int i = 0; i < 4; ++i) mu_s[i] = muf[o4 + i];
#pragma unroll
    for (int i = 0; i < 16; ++i) sig_s[i] = sigf[o16 + i];
    size_t ob = ((size_t)b * T_ + (T_ - 1)) * 20;
#pragma unroll
    for (int i = 0; i < 4; ++i) out[ob + i] = (float)mu_s[i];
#pragma unroll
    for (int i = 0; i < 16; ++i) out[ob + 4 + i] = (float)sig_s[i];
  }

  for (int t = T_ - 2; t >= 0; --t) {
    size_t o4 = ((size_t)t * B_ + b) * 4, o16 = ((size_t)t * B_ + b) * 16;
    size_t p4 = ((size_t)(t + 1) * B_ + b) * 4, p16 = ((size_t)(t + 1) * B_ + b) * 16;
    double mft[4], sft[16], mp1[4], sp1[16];
#pragma unroll
    for (int i = 0; i < 4; ++i) { mft[i] = muf[o4 + i]; mp1[i] = mup[p4 + i]; }
#pragma unroll
    for (int i = 0; i < 16; ++i) { sft[i] = sigf[o16 + i]; sp1[i] = sigp[p16 + i]; }

    const float* w1 = &wsm[((size_t)b * T_ + t + 1) * 3];
    double At1[16];
#pragma unroll
    for (int i = 0; i < 16; ++i)
      At1[i] = (double)(w1[0] * sA[0][i] + w1[1] * sA[1][i] + w1[2] * sA[2][i]);

    // Cholesky sp1 = L L^T
    double Lm[16];
#pragma unroll
    for (int i = 0; i < 16; ++i) Lm[i] = 0.0;
#pragma unroll
    for (int j = 0; j < 4; ++j) {
      double s = sp1[j * 4 + j];
#pragma unroll
      for (int k = 0; k < 4; ++k)
        if (k < j) s -= Lm[j * 4 + k] * Lm[j * 4 + k];
      double d = sqrt(s);
      Lm[j * 4 + j] = d;
      double invd = 1.0 / d;
#pragma unroll
      for (int i = 0; i < 4; ++i)
        if (i > j) {
          double s2 = sp1[i * 4 + j];
#pragma unroll
          for (int k = 0; k < 4; ++k)
            if (k < j) s2 -= Lm[i * 4 + k] * Lm[j * 4 + k];
          Lm[i * 4 + j] = s2 * invd;
        }
    }
    // Solve sp1 * X = At1  ->  X = inv(sp1) @ At1
    double X[16];
#pragma unroll
    for (int c = 0; c < 4; ++c) {
      double yv[4];
#pragma unroll
      for (int i = 0; i < 4; ++i) {
        double s = At1[i * 4 + c];
#pragma unroll
        for (int k = 0; k < 4; ++k)
          if (k < i) s -= Lm[i * 4 + k] * yv[k];
        yv[i] = s / Lm[i * 4 + i];
      }
#pragma unroll
      for (int i2 = 3; i2 >= 0; --i2) {
        double s = yv[i2];
#pragma unroll
        for (int k = 0; k < 4; ++k)
          if (k > i2) s -= Lm[k * 4 + i2] * X[k * 4 + c];
        X[i2 * 4 + c] = s / Lm[i2 * 4 + i2];
      }
    }
    // J = sig_f @ At1^T @ inv(sp1) = sft @ X^T  (sp1 symmetric)
    double J[16];
#pragma unroll
    for (int i = 0; i < 4; ++i)
#pragma unroll
      for (int j = 0; j < 4; ++j) {
        double s = 0;
#pragma unroll
        for (int l = 0; l < 4; ++l) s += sft[i * 4 + l] * X[j * 4 + l];
        J[i * 4 + j] = s;
      }
    double dm[4];
#pragma unroll
    for (int l = 0; l < 4; ++l) dm[l] = mu_s[l] - mp1[l];
    double mu_n[4];
#pragma unroll
    for (int i = 0; i < 4; ++i) {
      double s = mft[i];
#pragma unroll
      for (int l = 0; l < 4; ++l) s += J[i * 4 + l] * dm[l];
      mu_n[i] = s;
    }
    double DS[16];
#pragma unroll
    for (int i = 0; i < 16; ++i) DS[i] = sig_s[i] - sp1[i];
    double JD[16];
#pragma unroll
    for (int i = 0; i < 4; ++i)
#pragma unroll
      for (int j = 0; j < 4; ++j) {
        double s = 0;
#pragma unroll
        for (int l = 0; l < 4; ++l) s += J[i * 4 + l] * DS[l * 4 + j];
        JD[i * 4 + j] = s;
      }
    double sig_n[16];
#pragma unroll
    for (int i = 0; i < 4; ++i)
#pragma unroll
      for (int j = 0; j < 4; ++j) {
        double s = sft[i * 4 + j];
#pragma unroll
        for (int l = 0; l < 4; ++l) s += JD[i * 4 + l] * J[j * 4 + l];
        sig_n[i * 4 + j] = s;
      }
#pragma unroll
    for (int i = 0; i < 4; ++i) mu_s[i] = mu_n[i];
#pragma unroll
    for (int i = 0; i < 16; ++i) sig_s[i] = sig_n[i];

    size_t ob = ((size_t)b * T_ + t) * 20;
#pragma unroll
    for (int i = 0; i < 4; ++i) out[ob + i] = (float)mu_s[i];
#pragma unroll
    for (int i = 0; i < 16; ++i) out[ob + 4 + i] = (float)sig_s[i];
  }
}

// ---------------------------------------------------------------------------
extern "C" void kernel_launch(void* const* d_in, const int* in_sizes, int n_in,
                              void* d_out, int out_size, void* d_ws, size_t ws_size,
                              hipStream_t stream) {
  (void)in_sizes; (void)n_in; (void)out_size; (void)ws_size;
  const float* obs  = (const float*)d_in[0];
  const float* Wih0 = (const float*)d_in[1];
  const float* Whh0 = (const float*)d_in[2];
  const float* bih0 = (const float*)d_in[3];
  const float* bhh0 = (const float*)d_in[4];
  const float* Wih1 = (const float*)d_in[5];
  const float* Whh1 = (const float*)d_in[6];
  const float* bih1 = (const float*)d_in[7];
  const float* bhh1 = (const float*)d_in[8];
  const float* Walp = (const float*)d_in[9];
  const float* balp = (const float*)d_in[10];
  const float* a1   = (const float*)d_in[11];
  const float* Ak   = (const float*)d_in[12];
  const float* Ck   = (const float*)d_in[13];
  float* out = (float*)d_out;

  char* ws = (char*)d_ws;
  size_t off = 0;
  auto alloc = [&](size_t bytes) {
    size_t p = off;
    off += (bytes + 255) & ~(size_t)255;
    return p;
  };
  // xprojT: 16384 tiles x 13 ntiles x 32 lanes x 8 floats (reused both layers)
  float*    xprojT = (float*)(ws + alloc((size_t)(B_ / 16) * T_ * NT_ * 32 * 8 * 4));
  _Float16* h0A    = (_Float16*)(ws + alloc((size_t)(B_ / 16) * T_ * 1024 * 2));
  float*    wsm    = (float*)(ws + alloc((size_t)B_ * T_ * 3 * 4));
  double*   muf    = (double*)(ws + alloc((size_t)T_ * B_ * 4 * 8));
  double*   sigf   = (double*)(ws + alloc((size_t)T_ * B_ * 16 * 8));
  double*   mup    = (double*)(ws + alloc((size_t)T_ * B_ * 4 * 8));
  double*   sigp   = (double*)(ws + alloc((size_t)T_ * B_ * 16 * 8));

  {
    size_t total = (size_t)B_ * T_ * NP_;
    xproj0_kernel<<<(unsigned)((total + 255) / 256), 256, 0, stream>>>(
        obs, Wih0, bih0, bhh0, a1, xprojT);
  }
  lstm_rec_kernel<0><<<B_ / 16, 128, 0, stream>>>(Whh0, xprojT, h0A, nullptr, nullptr, nullptr);
  xproj1_gemm_kernel<<<(B_ / 16) * T_ / 4, 128, 0, stream>>>(h0A, Wih1, bih1, bhh1, xprojT);
  lstm_rec_kernel<1><<<B_ / 16, 128, 0, stream>>>(Whh1, xprojT, nullptr, Walp, balp, wsm);
  kalman_filter_kernel<<<B_ / 128, 128, 0, stream>>>(obs, wsm, Ak, Ck, muf, sigf, mup, sigp);
  kalman_smooth_kernel<<<B_ / 128, 128, 0, stream>>>(wsm, Ak, muf, sigf, mup, sigp, out);
}
